// ParallelLorentzEncoder_77214922047756
// MI455X (gfx1250) — compile-verified, matrix operand order checked by reading, not run on hardware
//
#include <hip/hip_runtime.h>
#include <math.h>

typedef __attribute__((ext_vector_type(16))) _Float16 v16h;
typedef __attribute__((ext_vector_type(8)))  float    v8f;

#define D_MODEL 64
#define D_INNER 128
#define D_STATE 16
#define DT_RANK 4
#define D_CONV  4
#define EMBED   32
#define BB      32
#define TT      2048
#define MM      (BB*TT)     /* 65536 rows */
#define XD_LD   48          /* padded x_proj output stride (36 valid) */
#define EPSF    1e-6f

// ---------------------------------------------------------------- embed
__global__ void k_embed(const float* __restrict__ x, const float* __restrict__ w,
                        const float* __restrict__ b, float* __restrict__ h) {
  int i = blockIdx.x * blockDim.x + threadIdx.x;
  if (i >= MM * D_MODEL) return;
  int d = i & (D_MODEL - 1);
  int row = i >> 6;
  h[i] = x[row] * w[d] + b[d];
}

// Load 8 contiguous floats as two b128 loads, convert to f16 into dst[off..off+7].
__device__ __forceinline__ void load8_cvt(const float* __restrict__ p, v16h& dst, int off) {
  const float4* q = (const float4*)p;
  float4 f0 = q[0];
  float4 f1 = q[1];
  dst[off + 0] = (_Float16)f0.x;  dst[off + 1] = (_Float16)f0.y;
  dst[off + 2] = (_Float16)f0.z;  dst[off + 3] = (_Float16)f0.w;
  dst[off + 4] = (_Float16)f1.x;  dst[off + 5] = (_Float16)f1.y;
  dst[off + 6] = (_Float16)f1.z;  dst[off + 7] = (_Float16)f1.w;
}
__device__ __forceinline__ void load8_cvt_scaled(const float* __restrict__ p, float m,
                                                 v16h& dst, int off) {
  const float4* q = (const float4*)p;
  float4 f0 = q[0];
  float4 f1 = q[1];
  dst[off + 0] = (_Float16)(f0.x * m);  dst[off + 1] = (_Float16)(f0.y * m);
  dst[off + 2] = (_Float16)(f0.z * m);  dst[off + 3] = (_Float16)(f0.w * m);
  dst[off + 4] = (_Float16)(f1.x * m);  dst[off + 5] = (_Float16)(f1.y * m);
  dst[off + 6] = (_Float16)(f1.z * m);  dst[off + 7] = (_Float16)(f1.w * m);
}

// ---------------------------------------------------------------- WMMA GEMM
// C[M,N] = A[M,K] * W[N,K]^T ; A row-major (lda=K), W row-major (ld=K).
// Each wave computes a 64x16 tile: 4 M-subtiles sharing one B fragment per
// K-step. K is a compile-time constant -> fully unrolled -> K/32*4 static WMMAs.
template <int K>
__global__ __launch_bounds__(128) void k_gemm_wmma(
    const float* __restrict__ A, const float* __restrict__ W,
    float* __restrict__ C, int ldc, int Nvalid) {
  const int wid  = threadIdx.x >> 5;
  const int lane = threadIdx.x & 31;
  const int ntiles = (Nvalid + 15) >> 4;
  const int ntile  = blockIdx.y * 4 + wid;
  if (ntile >= ntiles) return;              // wave-uniform: EXEC stays all-1s
  const int mbase = blockIdx.x << 6;        // 64 rows per wave
  const int nbase = ntile << 4;
  const int mr = lane & 15;                 // A row within subtile / B column
  const int hi = lane >> 4;                 // half-wave selector

  // Branch-free B row: clamp out-of-range column to a valid row, zero by mask.
  const int nrow = nbase + mr;
  const int nclamp = (nrow < Nvalid) ? nrow : (Nvalid - 1);
  const float bmask = (nrow < Nvalid) ? 1.0f : 0.0f;
  const float* __restrict__ Wrow = W + (size_t)nclamp * K;

  const float* __restrict__ Ar0 = A + (size_t)(mbase + mr) * K;

  v8f acc0 = {}, acc1 = {}, acc2 = {}, acc3 = {};
#pragma unroll
  for (int kb = 0; kb < K; kb += 32) {
    // Fragment K mapping: elems 0..7 <- K = kb + hi*8 + e (contiguous 8),
    //                     elems 8..15 <- K = kb + 16 + hi*8 + e (contiguous 8).
    v16h b;
    load8_cvt_scaled(Wrow + kb + hi * 8,      bmask, b, 0);
    load8_cvt_scaled(Wrow + kb + 16 + hi * 8, bmask, b, 8);
    v16h a0, a1, a2, a3;
    load8_cvt(Ar0 + 0 * 16 * K + kb + hi * 8,      a0, 0);
    load8_cvt(Ar0 + 0 * 16 * K + kb + 16 + hi * 8, a0, 8);
    load8_cvt(Ar0 + 1 * 16 * K + kb + hi * 8,      a1, 0);
    load8_cvt(Ar0 + 1 * 16 * K + kb + 16 + hi * 8, a1, 8);
    load8_cvt(Ar0 + 2 * 16 * K + kb + hi * 8,      a2, 0);
    load8_cvt(Ar0 + 2 * 16 * K + kb + 16 + hi * 8, a2, 8);
    load8_cvt(Ar0 + 3 * 16 * K + kb + hi * 8,      a3, 0);
    load8_cvt(Ar0 + 3 * 16 * K + kb + 16 + hi * 8, a3, 8);
    acc0 = __builtin_amdgcn_wmma_f32_16x16x32_f16(false, a0, false, b, (short)0, acc0, false, false);
    acc1 = __builtin_amdgcn_wmma_f32_16x16x32_f16(false, a1, false, b, (short)0, acc1, false, false);
    acc2 = __builtin_amdgcn_wmma_f32_16x16x32_f16(false, a2, false, b, (short)0, acc2, false, false);
    acc3 = __builtin_amdgcn_wmma_f32_16x16x32_f16(false, a3, false, b, (short)0, acc3, false, false);
  }

  const int col = nbase + (lane & 15);
  if (col < Nvalid) {
    // C layout: VGPR r -> M = r (lanes 0-15) / M = 8+r (lanes 16-31).
#pragma unroll
    for (int r = 0; r < 8; ++r) {
      int rowm = mbase + hi * 8 + r;
      C[(size_t)(rowm +  0) * ldc + col] = acc0[r];
      C[(size_t)(rowm + 16) * ldc + col] = acc1[r];
      C[(size_t)(rowm + 32) * ldc + col] = acc2[r];
      C[(size_t)(rowm + 48) * ldc + col] = acc3[r];
    }
  }
}

// ---------------------------------------------------------------- causal dwconv + silu
__global__ void k_conv_silu(const float* __restrict__ xz, const float* __restrict__ cw,
                            const float* __restrict__ cb, float* __restrict__ xin) {
  int i = blockIdx.x * blockDim.x + threadIdx.x;
  if (i >= MM * D_INNER) return;
  int d = i & (D_INNER - 1);
  int row = i >> 7;
  int t = row & (TT - 1);
  float v = cb[d];
#pragma unroll
  for (int k = 0; k < D_CONV; ++k) {
    int tt2 = t - (D_CONV - 1) + k;
    if (tt2 >= 0)
      v += xz[(size_t)(row - (D_CONV - 1) + k) * 256 + d] * cw[d * D_CONV + k];
  }
  xin[i] = v / (1.0f + __expf(-v));   // silu
}

// ---------------------------------------------------------------- selective scan (fused dt/softplus/gate)
__global__ __launch_bounds__(128) void k_scan(
    const float* __restrict__ xdbl, const float* __restrict__ xin,
    const float* __restrict__ xz,   const float* __restrict__ dt_proj,
    const float* __restrict__ dt_bias, const float* __restrict__ A_log,
    const float* __restrict__ Dp,   float* __restrict__ ybuf) {
  int b = blockIdx.x;
  int d = threadIdx.x;
  float negA[D_STATE];
#pragma unroll
  for (int s = 0; s < D_STATE; ++s) negA[s] = -__expf(A_log[d * D_STATE + s]);
  float dtp[DT_RANK];
#pragma unroll
  for (int r = 0; r < DT_RANK; ++r) dtp[r] = dt_proj[d * DT_RANK + r];
  float dtb = dt_bias[d], Dd = Dp[d];
  float h[D_STATE];
#pragma unroll
  for (int s = 0; s < D_STATE; ++s) h[s] = 0.0f;

  __shared__ float sB[D_STATE], sC[D_STATE], sdt[DT_RANK];
  for (int t = 0; t < TT; ++t) {
    size_t row = (size_t)b * TT + t;
    if (d < 16)                sB[d]       = xdbl[row * XD_LD + DT_RANK + d];
    else if (d < 32)           sC[d - 16]  = xdbl[row * XD_LD + DT_RANK + D_STATE + (d - 16)];
    else if (d < 32 + DT_RANK) sdt[d - 32] = xdbl[row * XD_LD + (d - 32)];
    __syncthreads();
    float u = xin[row * D_INNER + d];
    float pre = dtb;
#pragma unroll
    for (int r = 0; r < DT_RANK; ++r) pre += sdt[r] * dtp[r];
    float delta = (pre > 20.0f) ? pre : log1pf(__expf(pre));
    float y = 0.0f;
#pragma unroll
    for (int s = 0; s < D_STATE; ++s) {
      float dA = __expf(delta * negA[s]);
      h[s] = dA * h[s] + delta * sB[s] * u;
      y += h[s] * sC[s];
    }
    float yy = y + u * Dd;
    float z = xz[row * 256 + D_INNER + d];
    float g = z / (1.0f + __expf(-z));
    ybuf[row * D_INNER + d] = yy * g;
    __syncthreads();
  }
}

// ---------------------------------------------------------------- mean over T
__global__ void k_mean(const float* __restrict__ h, float* __restrict__ mh) {
  int b = blockIdx.x, d = threadIdx.x;
  float s = 0.0f;
  for (int t = 0; t < TT; ++t) s += h[((size_t)b * TT + t) * D_MODEL + d];
  mh[b * D_MODEL + d] = s * (1.0f / TT);
}

// ---------------------------------------------------------------- output head
__global__ void k_head(const float* __restrict__ mh, const float* __restrict__ ow,
                       const float* __restrict__ ob, float* __restrict__ z) {
  int e = threadIdx.x, b = threadIdx.y;
  float acc = ob[e];
  for (int d = 0; d < D_MODEL; ++d) acc += mh[b * D_MODEL + d] * ow[e * D_MODEL + d];
  z[b * EMBED + e] = acc;
}

// ---------------------------------------------------------------- Lorentz maps
__device__ inline float wsum(float v) {
#pragma unroll
  for (int off = 16; off >= 1; off >>= 1) v += __shfl_xor(v, off, 32);
  return v;
}
__device__ inline float d_expmap_projx(float u, int lane) {
  float c = (lane == 0) ? -u * u : u * u;
  float inner = wsum(c);
  float nom = sqrtf(fmaxf(inner, EPSF));
  float r = sinhf(nom) * u / fmaxf(nom, EPSF);
  float e = (lane == 0) ? coshf(nom) : r;
  float sp = (lane == 0) ? 0.0f : e;
  float ss = wsum(sp * sp);
  return (lane == 0) ? sqrtf(1.0f + ss) : e;
}
__device__ inline float d_logmap0(float h, int lane) {
  float a = __shfl(h, 0, 32);
  float ac = fmaxf(a, 1.0f + EPSF);
  float dist = logf(ac + sqrtf(ac * ac - 1.0f));
  float coef = dist / sqrtf(fmaxf(a * a - 1.0f, EPSF));
  return (lane == 0) ? 0.0f : coef * h;
}
__global__ __launch_bounds__(32) void k_hyper(float* out) {
  int b = blockIdx.x, i = threadIdx.x;
  float ut = out[0 * 1024 + b * 32 + i];
  float us = out[1 * 1024 + b * 32 + i];
  float ur = out[2 * 1024 + b * 32 + i];
  float hT = d_expmap_projx(ut, i);
  float hS = d_expmap_projx(us, i);
  float hR = d_expmap_projx(ur, i);
  out[3 * 1024 + b * 32 + i] = hT;
  out[4 * 1024 + b * 32 + i] = hS;
  out[5 * 1024 + b * 32 + i] = hR;
  float ct = d_logmap0(hT, i) + d_logmap0(hS, i) + d_logmap0(hR, i);
  out[6 * 1024 + b * 32 + i] = ct;
  out[7 * 1024 + b * 32 + i] = d_expmap_projx(ct, i);
}

// ---------------------------------------------------------------- launcher
extern "C" void kernel_launch(void* const* d_in, const int* in_sizes, int n_in,
                              void* d_out, int out_size, void* d_ws, size_t ws_size,
                              hipStream_t stream) {
  (void)out_size; (void)ws_size;
  float* out = (float*)d_out;
  float* ws  = (float*)d_ws;
  float* h    = ws;                                  // MM*64
  float* xz   = h    + (size_t)MM * D_MODEL;         // MM*256
  float* xin  = xz   + (size_t)MM * 256;             // MM*128
  float* xdbl = xin  + (size_t)MM * D_INNER;         // MM*48
  float* ybuf = xdbl + (size_t)MM * XD_LD;           // MM*128
  float* mh   = ybuf + (size_t)MM * D_INNER;         // 32*64

  // Detect pytree flattening order: sorted dict keys vs insertion order.
  bool sorted = true;
  if (n_in > 5 && in_sizes[5] == 16384) sorted = false;   // in_proj first => insertion order

  for (int enc = 0; enc < 3; ++enc) {
    const float* x = (const float*)d_in[enc];
    int base = 3 + enc * 31;
    auto P = [&](int i) { return (const float*)d_in[i]; };
    const float *in_w, *in_b, *out_w, *out_b;
    const float *ipj[3], *cw[3], *cb[3], *xpj[3], *dtp[3], *dtb[3], *alog[3], *Dp[3], *opj[3];
    if (sorted) {
      in_b = P(base + 0); in_w = P(base + 1);
      for (int l = 0; l < 3; ++l) {
        int lb = base + 2 + 9 * l;
        alog[l] = P(lb + 0); Dp[l]  = P(lb + 1); cb[l]  = P(lb + 2);
        cw[l]   = P(lb + 3); dtb[l] = P(lb + 4); dtp[l] = P(lb + 5);
        ipj[l]  = P(lb + 6); opj[l] = P(lb + 7); xpj[l] = P(lb + 8);
      }
      out_b = P(base + 29); out_w = P(base + 30);
    } else {
      in_w = P(base + 0); in_b = P(base + 1);
      for (int l = 0; l < 3; ++l) {
        int lb = base + 2 + 9 * l;
        ipj[l] = P(lb + 0); cw[l]  = P(lb + 1); cb[l]  = P(lb + 2);
        xpj[l] = P(lb + 3); dtp[l] = P(lb + 4); dtb[l] = P(lb + 5);
        alog[l]= P(lb + 6); Dp[l]  = P(lb + 7); opj[l] = P(lb + 8);
      }
      out_w = P(base + 29); out_b = P(base + 30);
    }

    k_embed<<<(MM * D_MODEL + 255) / 256, 256, 0, stream>>>(x, in_w, in_b, h);
    for (int l = 0; l < 3; ++l) {
      // xz = h @ in_proj^T  (M=65536, K=64, N=256): 64-row tiles -> grid.x = 1024
      k_gemm_wmma<64><<<dim3(MM / 64, 4), 128, 0, stream>>>(h, ipj[l], xz, 256, 256);
      k_conv_silu<<<(MM * D_INNER + 255) / 256, 256, 0, stream>>>(xz, cw[l], cb[l], xin);
      // xdbl = xin @ x_proj^T (K=128, N=36 padded to 48)
      k_gemm_wmma<128><<<dim3(MM / 64, 1), 128, 0, stream>>>(xin, xpj[l], xdbl, XD_LD, 36);
      k_scan<<<32, 128, 0, stream>>>(xdbl, xin, xz, dtp[l], dtb[l], alog[l], Dp[l], ybuf);
      // h = ybuf @ out_proj^T (K=128, N=64)
      k_gemm_wmma<128><<<dim3(MM / 64, 1), 128, 0, stream>>>(ybuf, opj[l], h, 64, 64);
    }
    k_mean<<<32, 64, 0, stream>>>(h, mh);
    k_head<<<1, dim3(32, 32), 0, stream>>>(mh, out_w, out_b, out + enc * 1024);
  }
  k_hyper<<<32, 32, 0, stream>>>(out);
}
